// BasicRNN_22368189677859
// MI455X (gfx1250) — matrix operand order repaired and checked by reading, not compiled, not run on hardware
//
#include <hip/hip_runtime.h>
#include <hip/hip_bf16.h>
#include <stdint.h>

// ---------- problem dims ----------
#define B_  32
#define T_  2048
#define D_  256
#define H_  1024
#define S_  1024
#define O_  256
#define BT_ (B_ * T_)

typedef unsigned short u16;
typedef __attribute__((ext_vector_type(16))) __bf16 v16bf;
typedef __attribute__((ext_vector_type(8)))  float  v8f;
typedef __attribute__((ext_vector_type(4)))  unsigned int v4u;
typedef __attribute__((ext_vector_type(8)))  int    v8i;
typedef __attribute__((ext_vector_type(4)))  int    v4i;

#if defined(__has_builtin)
#  if __has_builtin(__builtin_amdgcn_tensor_load_to_lds)
#    define HAVE_TDM 1
#  endif
#endif
#if __has_include(<hip/amd_detail/amd_gfx1250_TDM.h>)
#  define TDM_6ARG 1
#endif

struct Pair32 { uint4 a, b; };   // 32 bytes == one v16bf fragment

__device__ __forceinline__ u16 f2bf(float f) {
    unsigned u = __builtin_bit_cast(unsigned, f);
    unsigned r = u + 0x7FFFu + ((u >> 16) & 1u);      // round-to-nearest-even
    return (u16)(r >> 16);
}
__device__ __forceinline__ float bf2f(u16 h) {
    unsigned u = ((unsigned)h) << 16;
    return __builtin_bit_cast(float, u);
}

__device__ __forceinline__ v8f wmma_bf16(v16bf a, v16bf b, v8f c) {
    // (neg_a, A, neg_b, B, c_mod, C, reuse_a, reuse_b)
    return __builtin_amdgcn_wmma_f32_16x16x32_bf16(false, a, false, b, (short)0, c,
                                                   false, false);
}

// A fragment: 16x32 bf16.  lane<16: row M=lane, K k0..k0+15
//             lane>=16: row M=lane-16, K k0+16..k0+31
__device__ __forceinline__ v16bf frag_a(const u16* __restrict__ arow, int k0, int hi8) {
    Pair32 p;
    p.a = *(const uint4*)(arow + k0 + 2 * hi8);
    p.b = *(const uint4*)(arow + k0 + 2 * hi8 + 8);
    return __builtin_bit_cast(v16bf, p);
}
// B fragment: 32x16 bf16. lane n<16: col n, K k0..k0+15 (32 contiguous bytes of
// the weight row); lane n+16: col n, K k0+16..k0+31.
__device__ __forceinline__ v16bf frag_b(const u16* __restrict__ wrow_at_k) {
    Pair32 p;
    p.a = *(const uint4*)(wrow_at_k);
    p.b = *(const uint4*)(wrow_at_k + 8);
    return __builtin_bit_cast(v16bf, p);
}

// ---------------------------------------------------------------------------
// fp32 -> bf16 cast
// ---------------------------------------------------------------------------
__global__ void cast_f32_bf16(const float* __restrict__ in, u16* __restrict__ out,
                              long long n) {
    long long i = (long long)blockIdx.x * blockDim.x + threadIdx.x;
    long long stride = (long long)gridDim.x * blockDim.x;
    for (; i < n; i += stride) out[i] = f2bf(in[i]);
}

__global__ void init_state(u16* h0, u16* h1, unsigned* ctr) {
    int i = blockIdx.x * blockDim.x + threadIdx.x;
    if (i < B_ * S_) { h0[i] = 0; h1[i] = 0; }
    if (i == 0) *ctr = 0;
}

// ---------------------------------------------------------------------------
// WMMA GEMM:  out[M,N] = A[M,K](bf16) * W[N,K]^T(bf16) + bias
// grid = (N/64, M/256), block = 256 (8 waves).
// Each wave computes a 32x64 tile (2 M-tiles x 4 N-tiles): per k-step,
// 4 A-loads + 8 B-loads feed 8 WMMAs (B fragments reused across both M tiles).
// ---------------------------------------------------------------------------
__global__ __launch_bounds__(256)
void wmma_gemm(const u16* __restrict__ A, const u16* __restrict__ W,
               const float* __restrict__ bias,
               u16* __restrict__ outBf, float* __restrict__ outF,
               int N, int K, int tanh_out) {
    const int lane = threadIdx.x & 31;
    const int wave = threadIdx.x >> 5;
    const int ln   = lane & 15;
    const int hi8  = (lane >> 4) * 8;
    const int hi16 = (lane >> 4) * 16;

    const int m0 = blockIdx.y * 256 + wave * 32;
    const int nb = blockIdx.x * 64;

    const u16* arow0 = A + (size_t)(m0 +  0 + ln) * K;
    const u16* arow1 = A + (size_t)(m0 + 16 + ln) * K;
    const u16* w0 = W + (size_t)(nb +  0 + ln) * K + hi16;
    const u16* w1 = W + (size_t)(nb + 16 + ln) * K + hi16;
    const u16* w2 = W + (size_t)(nb + 32 + ln) * K + hi16;
    const u16* w3 = W + (size_t)(nb + 48 + ln) * K + hi16;

    v8f acc[2][4];
#pragma unroll
    for (int mt = 0; mt < 2; ++mt)
#pragma unroll
        for (int nt = 0; nt < 4; ++nt)
            acc[mt][nt] = (v8f){0, 0, 0, 0, 0, 0, 0, 0};

#pragma unroll 4
    for (int k0 = 0; k0 < K; k0 += 32) {
        v16bf a0 = frag_a(arow0, k0, hi8);
        v16bf a1 = frag_a(arow1, k0, hi8);
        v16bf b0 = frag_b(w0 + k0);
        v16bf b1 = frag_b(w1 + k0);
        v16bf b2 = frag_b(w2 + k0);
        v16bf b3 = frag_b(w3 + k0);
        acc[0][0] = wmma_bf16(a0, b0, acc[0][0]);
        acc[1][0] = wmma_bf16(a1, b0, acc[1][0]);
        acc[0][1] = wmma_bf16(a0, b1, acc[0][1]);
        acc[1][1] = wmma_bf16(a1, b1, acc[1][1]);
        acc[0][2] = wmma_bf16(a0, b2, acc[0][2]);
        acc[1][2] = wmma_bf16(a1, b2, acc[1][2]);
        acc[0][3] = wmma_bf16(a0, b3, acc[0][3]);
        acc[1][3] = wmma_bf16(a1, b3, acc[1][3]);
    }

#pragma unroll
    for (int mt = 0; mt < 2; ++mt) {
#pragma unroll
        for (int nt = 0; nt < 4; ++nt) {
            const int n = nb + nt * 16 + ln;
            const float bv = bias[n];
#pragma unroll
            for (int r = 0; r < 8; ++r) {
                const int m = m0 + mt * 16 + ((lane < 16) ? r : 8 + r);
                const float v = acc[mt][nt][r] + bv;
                if (tanh_out) outF[(size_t)m * N + n] = tanhf(v);
                else          outBf[(size_t)m * N + n] = f2bf(v);
            }
        }
    }
}

// ---------------------------------------------------------------------------
// TDM: 2-D tile (tile0 x tile1 rows) of bf16 from global -> LDS
// ---------------------------------------------------------------------------
#ifdef HAVE_TDM
__device__ __forceinline__ void tdm_load_2d(unsigned lds_addr, unsigned long long ga,
                                            unsigned dim0, unsigned dim1,
                                            unsigned tile0, unsigned tile1,
                                            unsigned long long stride0) {
    v4u g0;
    g0.x = 1u;                                           // count=1, user descriptor
    g0.y = lds_addr;                                     // LDS byte address
    g0.z = (unsigned)(ga & 0xFFFFFFFFu);                 // global_addr[31:0]
    g0.w = (unsigned)((ga >> 32) & 0x01FFFFFFu) | (2u << 30);  // addr hi + type=2
    v8i g1 = {
        (int)(1u << 16),                                           // data_size=2B
        (int)((dim0 & 0xFFFFu) << 16),                             // tensor_dim0 lo
        (int)((dim0 >> 16) | ((dim1 & 0xFFFFu) << 16)),            // dim0 hi | dim1 lo
        (int)((dim1 >> 16) | ((tile0 & 0xFFFFu) << 16)),           // dim1 hi | tile_dim0
        (int)(tile1 & 0xFFFFu),                                    // tile_dim1 (tile2=0)
        (int)(unsigned)(stride0 & 0xFFFFFFFFu),                    // dim0_stride lo
        (int)((stride0 >> 32) & 0xFFFFu),                          // dim0_stride hi
        0 };
    v4i g2 = {0, 0, 0, 0};
    v4i g3 = {0, 0, 0, 0};
#ifdef TDM_6ARG
    v8i g4 = {0, 0, 0, 0, 0, 0, 0, 0};
    __builtin_amdgcn_tensor_load_to_lds(g0, g1, g2, g3, g4, 0);
#else
    __builtin_amdgcn_tensor_load_to_lds(g0, g1, g2, g3, 0);
#endif
}
#endif

// ---------------------------------------------------------------------------
// Persistent GRU recurrence.
// grid = S/16 = 64 WGs; block = 64 (2 waves). WG owns columns [s0, s0+16) of S
// across all 3 gates; wave w owns batch rows [16w, 16w+16).
// w_hh slice (3 gates x 16 rows x 1024 K, bf16 = 96 KB) lives in LDS.
// ---------------------------------------------------------------------------
__global__ __launch_bounds__(64)
void gru_persistent(const u16* __restrict__ igates,   // [BT, 3S] bf16
                    const u16* __restrict__ whh,      // [3S, S]  bf16
                    const float* __restrict__ b_n,    // [S]
                    u16* __restrict__ h0buf, u16* __restrict__ h1buf,
                    u16* __restrict__ states,         // [BT, S] bf16
                    unsigned* __restrict__ ctr) {
    __shared__ __align__(16) u16 wtile[3 * 16 * S_];   // 96 KB of 320 KB WGP LDS

    const int lane = threadIdx.x & 31;
    const int wave = threadIdx.x >> 5;
    const int ln   = lane & 15;
    const int hi8  = (lane >> 4) * 8;
    const int hi16 = (lane >> 4) * 16;
    const int s0   = blockIdx.x * 16;
    const int m0   = wave * 16;
    const int nWG  = gridDim.x;

    // ---- stage the 3 gate row-blocks of w_hh into LDS ----
#ifdef HAVE_TDM
    if (wave == 0) {
#pragma unroll
        for (int g = 0; g < 3; ++g) {
            unsigned lds_off = (unsigned)(unsigned long long)(uintptr_t)&wtile[g * 16 * S_];
            unsigned long long ga =
                (unsigned long long)(uintptr_t)(whh + (size_t)(g * S_ + s0) * S_);
            tdm_load_2d(lds_off, ga, /*dim0=*/S_, /*dim1=*/16,
                        /*tile0=*/S_, /*tile1=*/16, /*stride0=*/S_);
        }
        __builtin_amdgcn_s_wait_tensorcnt(0);
    }
#else
    for (int i = threadIdx.x; i < 3 * 16 * (S_ / 8); i += blockDim.x) {
        const int g   = i / (16 * (S_ / 8));
        const int rem = i % (16 * (S_ / 8));
        const int row = rem / (S_ / 8);
        const int kq  = rem % (S_ / 8);
        *(uint4*)&wtile[(g * 16 + row) * S_ + kq * 8] =
            *(const uint4*)(whh + (size_t)(g * S_ + s0 + row) * S_ + kq * 8);
    }
#endif
    __syncthreads();

    const int wb0 = (0 * 16 + ln) * S_ + hi16;
    const int wb1 = (1 * 16 + ln) * S_ + hi16;
    const int wb2 = (2 * 16 + ln) * S_ + hi16;
    const int n   = s0 + ln;
    const float bn = b_n[n];

    u16* hbuf[2] = { h0buf, h1buf };

    for (int t = 0; t < T_; ++t) {
        const u16* hc = hbuf[t & 1];
        u16*       hx = hbuf[(t + 1) & 1];

        v8f ar = {0,0,0,0,0,0,0,0}, az = {0,0,0,0,0,0,0,0}, an = {0,0,0,0,0,0,0,0};
        const u16* hrow = hc + (size_t)(m0 + ln) * S_;

#pragma unroll 4
        for (int k0 = 0; k0 < S_; k0 += 32) {
            v16bf a = frag_a(hrow, k0, hi8);
            Pair32 p0, p1, p2;
            p0.a = *(const uint4*)&wtile[wb0 + k0];
            p0.b = *(const uint4*)&wtile[wb0 + k0 + 8];
            p1.a = *(const uint4*)&wtile[wb1 + k0];
            p1.b = *(const uint4*)&wtile[wb1 + k0 + 8];
            p2.a = *(const uint4*)&wtile[wb2 + k0];
            p2.b = *(const uint4*)&wtile[wb2 + k0 + 8];
            ar = wmma_bf16(a, __builtin_bit_cast(v16bf, p0), ar);
            az = wmma_bf16(a, __builtin_bit_cast(v16bf, p1), az);
            an = wmma_bf16(a, __builtin_bit_cast(v16bf, p2), an);
        }

        // pointwise GRU update for this wave's 16 rows x 16 cols
#pragma unroll
        for (int r = 0; r < 8; ++r) {
            const int bidx = m0 + ((lane < 16) ? r : 8 + r);
            const size_t gbase = ((size_t)bidx * T_ + t) * (3 * S_);
            const float ir = bf2f(igates[gbase + 0 * S_ + n]);
            const float iz = bf2f(igates[gbase + 1 * S_ + n]);
            const float in_ = bf2f(igates[gbase + 2 * S_ + n]);
            const float rg = 1.f / (1.f + __expf(-(ir + ar[r])));
            const float zg = 1.f / (1.f + __expf(-(iz + az[r])));
            const float ng = tanhf(in_ + rg * (an[r] + bn));
            const float hold = bf2f(hc[(size_t)bidx * S_ + n]);
            const float hnew = ng + zg * (hold - ng);
            const u16 hb = f2bf(hnew);
            hx[(size_t)bidx * S_ + n] = hb;
            states[((size_t)bidx * T_ + t) * S_ + n] = hb;
        }

        // -------- grid-wide barrier (all 64 WGs co-resident) --------
        __threadfence();
        __syncthreads();
        if (threadIdx.x == 0) {
            __hip_atomic_fetch_add(ctr, 1u, __ATOMIC_ACQ_REL, __HIP_MEMORY_SCOPE_AGENT);
            const unsigned target = (unsigned)(t + 1) * (unsigned)nWG;
            while (__hip_atomic_load(ctr, __ATOMIC_ACQUIRE, __HIP_MEMORY_SCOPE_AGENT)
                   < target) {
                __builtin_amdgcn_s_sleep(1);
            }
        }
        __syncthreads();
    }
}

// ---------------------------------------------------------------------------
// host-side orchestration
// ---------------------------------------------------------------------------
extern "C" void kernel_launch(void* const* d_in, const int* in_sizes, int n_in,
                              void* d_out, int out_size, void* d_ws, size_t ws_size,
                              hipStream_t stream) {
    const float* x     = (const float*)d_in[0];
    const float* enc_w = (const float*)d_in[1];
    const float* enc_b = (const float*)d_in[2];
    const float* w_ih  = (const float*)d_in[3];
    const float* w_hh  = (const float*)d_in[4];
    const float* bgate = (const float*)d_in[5];
    const float* b_n   = (const float*)d_in[6];
    const float* dec_w = (const float*)d_in[7];
    const float* dec_b = (const float*)d_in[8];
    float* out = (float*)d_out;

    char* ws = (char*)d_ws;
    size_t off = 0;
    auto alloc = [&](size_t bytes) -> char* {
        char* p = ws + off;
        off = (off + bytes + 255) & ~(size_t)255;
        return p;
    };
    unsigned* ctr   = (unsigned*)alloc(256);
    u16* h0         = (u16*)alloc((size_t)B_ * S_ * 2);
    u16* h1         = (u16*)alloc((size_t)B_ * S_ * 2);
    u16* x_bf       = (u16*)alloc((size_t)BT_ * D_ * 2);
    u16* encw_bf    = (u16*)alloc((size_t)H_ * D_ * 2);
    u16* wih_bf     = (u16*)alloc((size_t)3 * S_ * H_ * 2);
    u16* whh_bf     = (u16*)alloc((size_t)3 * S_ * S_ * 2);
    u16* decw_bf    = (u16*)alloc((size_t)O_ * S_ * 2);
    u16* xe_bf      = (u16*)alloc((size_t)BT_ * H_ * 2);
    u16* ig_bf      = (u16*)alloc((size_t)BT_ * 3 * S_ * 2);
    u16* states_bf  = (u16*)alloc((size_t)BT_ * S_ * 2);

    init_state<<<(B_ * S_ + 255) / 256, 256, 0, stream>>>(h0, h1, ctr);

    auto cast = [&](const float* src, u16* dst, long long n) {
        int blocks = (int)((n + 255) / 256);
        if (blocks > 4096) blocks = 4096;
        cast_f32_bf16<<<blocks, 256, 0, stream>>>(src, dst, n);
    };
    cast(x,     x_bf,    (long long)BT_ * D_);
    cast(enc_w, encw_bf, (long long)H_ * D_);
    cast(w_ih,  wih_bf,  (long long)3 * S_ * H_);
    cast(w_hh,  whh_bf,  (long long)3 * S_ * S_);
    cast(dec_w, decw_bf, (long long)O_ * S_);

    dim3 blk(256);
    // encoder: xe[BT,H] = x[BT,D] * enc_w[H,D]^T + enc_b
    wmma_gemm<<<dim3(H_ / 64, BT_ / 256), blk, 0, stream>>>(
        x_bf, encw_bf, enc_b, xe_bf, nullptr, H_, D_, 0);
    // igates[BT,3S] = xe[BT,H] * w_ih[3S,H]^T + b
    wmma_gemm<<<dim3(3 * S_ / 64, BT_ / 256), blk, 0, stream>>>(
        xe_bf, wih_bf, bgate, ig_bf, nullptr, 3 * S_, H_, 0);
    // sequential GRU over T
    gru_persistent<<<dim3(S_ / 16), dim3(64), 0, stream>>>(
        ig_bf, whh_bf, b_n, h0, h1, states_bf, ctr);
    // decoder: out[BT,O] = tanh(states[BT,S] * dec_w[O,S]^T + dec_b)
    wmma_gemm<<<dim3(O_ / 64, BT_ / 256), blk, 0, stream>>>(
        states_bf, decw_bf, dec_b, nullptr, out, O_, S_, 1);
}
